// AGCA_73581379715677
// MI455X (gfx1250) — compile-verified, matrix operand order checked by reading, not run on hardware
//
#include <hip/hip_runtime.h>
#include <hip/hip_bf16.h>

typedef __attribute__((ext_vector_type(16))) _Float16 v16h;
typedef __attribute__((ext_vector_type(8)))  float    v8f;
typedef __attribute__((ext_vector_type(4)))  float    v4f;

#define B_    64
#define CIN_  512
#define HIDE_ 128
#define OP_   512
#define HW_   4096

// ISA 7.12.2: 16-bit A-matrix 16x32 layout. Element e (0..15) of a lane maps to
// K = (e<8 ? e : e+8) + (lane>=16 ? 8 : 0). Row (A) / Col (B) = lane % 16.
__device__ __forceinline__ int kmap(int e, int lhi) {
  int k = (e < 8) ? e : (e + 8);
  return k + (lhi ? 8 : 0);
}

// ---------------------------------------------------------------------------
// Kernel 1: global average pool over H*W = 4096, one wave32 per (b, c) channel.
// Pure HBM streaming: 536 MB of x, non-temporal 128-bit loads.
// ---------------------------------------------------------------------------
__global__ __launch_bounds__(256) void agca_pool(const float* __restrict__ x,
                                                 float* __restrict__ y) {
  const int wave = threadIdx.x >> 5;
  const int lane = threadIdx.x & 31;
  const int ch   = blockIdx.x * 8 + wave;              // 0..32767
  const v4f* p = (const v4f*)(x + (size_t)ch * HW_) + lane;
  float s = 0.0f;
#pragma unroll
  for (int i = 0; i < 32; ++i) {
    v4f v = __builtin_nontemporal_load(p + i * 32);    // 128 floats per iter per wave
    s += v.x + v.y + v.z + v.w;
  }
#pragma unroll
  for (int off = 16; off > 0; off >>= 1)
    s += __shfl_xor(s, off, 32);
  if (lane == 0) y[ch] = s * (1.0f / (float)HW_);
}

// ---------------------------------------------------------------------------
// Kernel 2: entire tail in one workgroup (8 waves), WMMA f16->f32 GEMMs.
//   y1 = y @ W1^T            [64x512]@[512x128]
//   A1 = softmax(w2*y1)      per batch row
//   y2 = y1*A1 + y1 @ A2     [64x128]@[128x128]
//   y3 = relu(w3*y2)
//   out = sigmoid(y3 @ W4^T) [64x128]@[128x512]
// ---------------------------------------------------------------------------
__global__ __launch_bounds__(256) void agca_tail(
    const float* __restrict__ y,    // [64,512] pooled means (in d_ws)
    const float* __restrict__ W1,   // [128,512]
    const float* __restrict__ A2,   // [128,128]
    const float* __restrict__ w2p,
    const float* __restrict__ w3p,
    const float* __restrict__ W4,   // [512,128]
    float* __restrict__ out)        // [64,512]
{
  __shared__ float     y1s[B_ * HIDE_];   // 32 KB
  __shared__ _Float16  a1s[B_ * HIDE_];   // 16 KB
  __shared__ _Float16  y3s[B_ * HIDE_];   // 16 KB

  const int tid  = threadIdx.x;
  const int wave = tid >> 5;
  const int lane = tid & 31;
  const int l15  = lane & 15;
  const int lhi  = (lane >> 4) & 1;

  // ---- GEMM1: y1[64,128] = y @ W1^T, tiles: 4 M x 8 N, K = 512 ----
  for (int t = wave; t < 32; t += 8) {
    const int mt = t >> 3, nt = t & 7;
    const int row = mt * 16 + l15;   // batch row for A
    const int col = nt * 16 + l15;   // hidden col for B
    v8f acc = {};
    for (int kb = 0; kb < CIN_; kb += 32) {
      v16h a, b;
#pragma unroll
      for (int e = 0; e < 16; ++e) {
        const int k = kb + kmap(e, lhi);
        a[e] = (_Float16)y[row * CIN_ + k];
        b[e] = (_Float16)W1[col * CIN_ + k];   // B(K=c, N=h) = W1[h, c]
      }
      acc = __builtin_amdgcn_wmma_f32_16x16x32_f16(false, a, false, b,
                                                   (short)0, acc, false, false);
    }
    const int Mb = mt * 16 + lhi * 8;
#pragma unroll
    for (int r = 0; r < 8; ++r)
      y1s[(Mb + r) * HIDE_ + col] = acc[r];
  }
  __syncthreads();

  // ---- softmax over hidden dim (64 rows; one thread per row) ----
  const float w2 = w2p[0];
  const float w3 = w3p[0];
  if (tid < B_) {
    float m = -3.4e38f;
    for (int h = 0; h < HIDE_; ++h)
      m = fmaxf(m, w2 * y1s[tid * HIDE_ + h]);
    float s = 0.0f;
    for (int h = 0; h < HIDE_; ++h)
      s += expf(w2 * y1s[tid * HIDE_ + h] - m);
    const float inv = 1.0f / s;
    for (int h = 0; h < HIDE_; ++h)
      a1s[tid * HIDE_ + h] =
          (_Float16)(expf(w2 * y1s[tid * HIDE_ + h] - m) * inv);
  }
  __syncthreads();

  // ---- GEMM2: y1 @ A2, plus diagonal y1*A1, relu(w3*..) -> y3 (f16) ----
  for (int t = wave; t < 32; t += 8) {
    const int mt = t >> 3, nt = t & 7;
    const int row = mt * 16 + l15;
    const int col = nt * 16 + l15;
    v8f acc = {};
    for (int kb = 0; kb < HIDE_; kb += 32) {
      v16h a, b;
#pragma unroll
      for (int e = 0; e < 16; ++e) {
        const int k = kb + kmap(e, lhi);
        a[e] = (_Float16)y1s[row * HIDE_ + k];
        b[e] = (_Float16)A2[k * HIDE_ + col];  // B(K=h, N=k) = A2[h, k]
      }
      acc = __builtin_amdgcn_wmma_f32_16x16x32_f16(false, a, false, b,
                                                   (short)0, acc, false, false);
    }
    const int Mb = mt * 16 + lhi * 8;
#pragma unroll
    for (int r = 0; r < 8; ++r) {
      const int M = Mb + r;
      float v = acc[r] + y1s[M * HIDE_ + col] * (float)a1s[M * HIDE_ + col];
      v = w3 * v;
      y3s[M * HIDE_ + col] = (_Float16)fmaxf(v, 0.0f);
    }
  }
  __syncthreads();

  // ---- GEMM3: out[64,512] = sigmoid(y3 @ W4^T), tiles: 4 M x 32 N ----
  for (int t = wave; t < 128; t += 8) {
    const int mt = t >> 5, nt = t & 31;
    const int row = mt * 16 + l15;
    const int col = nt * 16 + l15;   // output channel o
    v8f acc = {};
    for (int kb = 0; kb < HIDE_; kb += 32) {
      v16h a, b;
#pragma unroll
      for (int e = 0; e < 16; ++e) {
        const int k = kb + kmap(e, lhi);
        a[e] = y3s[row * HIDE_ + k];
        b[e] = (_Float16)W4[col * HIDE_ + k];  // B(K=h, N=o) = W4[o, h]
      }
      acc = __builtin_amdgcn_wmma_f32_16x16x32_f16(false, a, false, b,
                                                   (short)0, acc, false, false);
    }
    const int Mb = mt * 16 + lhi * 8;
#pragma unroll
    for (int r = 0; r < 8; ++r) {
      const int M = Mb + r;
      out[M * OP_ + col] = 1.0f / (1.0f + expf(-acc[r]));
    }
  }
}

extern "C" void kernel_launch(void* const* d_in, const int* in_sizes, int n_in,
                              void* d_out, int out_size, void* d_ws, size_t ws_size,
                              hipStream_t stream) {
  (void)in_sizes; (void)n_in; (void)out_size; (void)ws_size;
  const float* x  = (const float*)d_in[0];   // [64,512,64,64]
  const float* W1 = (const float*)d_in[1];   // [128,512]
  const float* A2 = (const float*)d_in[2];   // [128,128]
  const float* w2 = (const float*)d_in[3];   // scalar
  const float* w3 = (const float*)d_in[4];   // scalar
  const float* W4 = (const float*)d_in[5];   // [512,128]
  float* y = (float*)d_ws;                   // [64,512] pooled means (128 KB)

  agca_pool<<<(B_ * CIN_) / 8, 256, 0, stream>>>(x, y);
  agca_tail<<<1, 256, 0, stream>>>(y, W1, A2, w2, w3, W4, (float*)d_out);
}
